// VQ_Layer_28973849379183
// MI455X (gfx1250) — matrix operand order repaired and checked by reading, not compiled, not run on hardware
//
#include <hip/hip_runtime.h>
#include <hip/hip_bf16.h>
#include <stdint.h>

// ---------------------------------------------------------------------------
// VQ layer: inputs [32,64,64,64] f32 -> N=131072 rows x D=64; codebook 1024x64
// Distance GEMM via v_wmma_f32_16x16x32_bf16 (f32 accumulate), accumulator
// pre-loaded with -0.5*|b|^2 so argmin(dist) == argmax(accumulator).
// Codebook staged to LDS with global_load_async_to_lds (ASYNCcnt).
// ---------------------------------------------------------------------------

typedef __attribute__((ext_vector_type(16))) __bf16 v16bf;
typedef __attribute__((ext_vector_type(8)))  float  v8f;
typedef int v4i __attribute__((vector_size(16)));

#define DDIM    64
#define KCODES  1024
#define NROWS   131072
#define NELEMS  8388608      // NROWS * DDIM
#define EPSF    1e-10f

// workspace layout (byte offsets)
#define WS_BF16  0           // uint32[32768] : 1024 x 64 bf16 packed (128 KB)
#define WS_BSQ   131072      // float [1024]  : -0.5*|b|^2 per code
#define WS_CNT   135168      // uint32[1024]  : code usage histogram
#define WS_PART  139264      // float [4096]  : per-wave squared-error partials

#define GLB __attribute__((address_space(1)))
#define LDS __attribute__((address_space(3)))

__device__ __forceinline__ uint32_t pack_bf16(float a, float b) {
  __bf16 x = (__bf16)a, y = (__bf16)b;             // RNE conversions
  uint16_t lo = __builtin_bit_cast(uint16_t, x);
  uint16_t hi = __builtin_bit_cast(uint16_t, y);
  return ((uint32_t)hi << 16) | (uint32_t)lo;
}

union Frag { uint32_t u[8]; uint4 q[2]; v16bf v; };

// ---------------------------------------------------------------------------
// Kernel 0: -0.5*|b|^2 per code, bf16 codebook, zero histogram
// ---------------------------------------------------------------------------
__global__ void vq_prep(const float* __restrict__ emb,
                        uint32_t* __restrict__ embbf,
                        float* __restrict__ bsqh,
                        uint32_t* __restrict__ counts) {
  int k = blockIdx.x * blockDim.x + threadIdx.x;
  if (k >= KCODES) return;
  const float4* r = (const float4*)(emb + (size_t)k * DDIM);
  uint32_t* dst = embbf + (size_t)k * (DDIM / 2);
  float s = 0.f;
  #pragma unroll
  for (int j = 0; j < DDIM / 4; ++j) {
    float4 x = r[j];
    s += x.x * x.x + x.y * x.y + x.z * x.z + x.w * x.w;
    dst[j * 2]     = pack_bf16(x.x, x.y);
    dst[j * 2 + 1] = pack_bf16(x.z, x.w);
  }
  bsqh[k] = -0.5f * s;
  counts[k] = 0u;
}

// ---------------------------------------------------------------------------
// Kernel 1: distances via WMMA, argmax, gather, loss partials
// 256 threads = 8 waves; wave handles 32 rows (2 A-tiles); block = 256 rows.
// ---------------------------------------------------------------------------
__launch_bounds__(256)
__global__ void vq_main(const float* __restrict__ inputs,
                        const float* __restrict__ emb,
                        const uint32_t* __restrict__ embbf,
                        const float* __restrict__ bsqh,
                        uint32_t* __restrict__ counts,
                        float* __restrict__ partials,
                        float* __restrict__ out) {
  __shared__ uint32_t ldsB[256 * 36];   // 256 codes x 64 bf16, 144B padded rows
  __shared__ float    ldsBsq[256];      // -0.5*|b|^2 tile
  __shared__ int      widx[8][32];      // per-wave chosen code per row

  const int tid  = threadIdx.x;
  const int wid  = tid >> 5;
  const int lane = tid & 31;
  const int n16  = lane & 15;
  const int half = lane >> 4;
  const int rowbase = blockIdx.x * 256 + wid * 32;

  // ---- Load A fragments (2 tiles x 16 rows x 64, bf16) into wave registers.
  // 16-bit A 16x32 layout: lane m = lane&15; v0..3 hold K = 8*half + {0..7},
  // v4..7 hold K = 16 + 8*half + {0..7}.  Fragment f adds 32 to K.
  Frag a[2][2];
  #pragma unroll
  for (int t = 0; t < 2; ++t) {
    const float* rp = inputs + (size_t)(rowbase + t * 16 + n16) * DDIM;
    const int kb = 8 * half;
    #pragma unroll
    for (int f = 0; f < 2; ++f) {
      const float* p = rp + f * 32 + kb;
      float4 x0 = *(const float4*)(p);
      float4 x1 = *(const float4*)(p + 4);
      float4 x2 = *(const float4*)(p + 16);
      float4 x3 = *(const float4*)(p + 20);
      a[t][f].u[0] = pack_bf16(x0.x, x0.y); a[t][f].u[1] = pack_bf16(x0.z, x0.w);
      a[t][f].u[2] = pack_bf16(x1.x, x1.y); a[t][f].u[3] = pack_bf16(x1.z, x1.w);
      a[t][f].u[4] = pack_bf16(x2.x, x2.y); a[t][f].u[5] = pack_bf16(x2.z, x2.w);
      a[t][f].u[6] = pack_bf16(x3.x, x3.y); a[t][f].u[7] = pack_bf16(x3.z, x3.w);
    }
  }

  float maxv[16];
  int   maxi[16];
  #pragma unroll
  for (int s = 0; s < 16; ++s) { maxv[s] = -3.4e38f; maxi[s] = 0; }

  // ---- Sweep codebook in 256-code LDS tiles
  for (int kt = 0; kt < 4; ++kt) {
    const int cbase = kt * 256;
    // cooperative async fill: one code row (128 B) + one bsq value per thread
#if __has_builtin(__builtin_amdgcn_global_load_async_to_lds_b128) && \
    __has_builtin(__builtin_amdgcn_global_load_async_to_lds_b32)  && \
    __has_builtin(__builtin_amdgcn_s_wait_asynccnt)
    {
      const uint32_t* src = embbf + (size_t)(cbase + tid) * 32;
      uint32_t* dst = ldsB + tid * 36;
      #pragma unroll
      for (int j = 0; j < 8; ++j)
        __builtin_amdgcn_global_load_async_to_lds_b128(
            (GLB v4i*)(src + j * 4), (LDS v4i*)(dst + j * 4), 0, 0);
      __builtin_amdgcn_global_load_async_to_lds_b32(
          (GLB int*)(bsqh + cbase + tid), (LDS int*)(&ldsBsq[tid]), 0, 0);
      __builtin_amdgcn_s_wait_asynccnt(0);
    }
#else
    {
      const uint4* src = (const uint4*)(embbf + (size_t)(cbase + tid) * 32);
      uint4 t0 = src[0], t1 = src[1], t2 = src[2], t3 = src[3];
      uint4 t4 = src[4], t5 = src[5], t6 = src[6], t7 = src[7];
      float bv = bsqh[cbase + tid];
      uint32_t* dst = ldsB + tid * 36;
      ((uint4*)dst)[0] = t0; *(uint4*)(dst + 4)  = t1;
      *(uint4*)(dst + 8)  = t2; *(uint4*)(dst + 12) = t3;
      *(uint4*)(dst + 16) = t4; *(uint4*)(dst + 20) = t5;
      *(uint4*)(dst + 24) = t6; *(uint4*)(dst + 28) = t7;
      ldsBsq[tid] = bv;
    }
#endif
    __syncthreads();

    #pragma unroll 2
    for (int nt = 0; nt < 16; ++nt) {
      // 16-bit B 32x16 layout: lane col n = lane&15; v0..3 hold K = 8*half + {0..7},
      // v4..7 hold K = 16 + 8*half + {0..7}; fragment f adds 32 to K (= +16 dwords in row).
      const int col = cbase + nt * 16 + n16;
      const uint32_t* bp = ldsB + (nt * 16 + n16) * 36 + 4 * half;
      Frag b0, b1;
      b0.q[0] = *(const uint4*)(bp);
      b0.q[1] = *(const uint4*)(bp + 8);
      b1.q[0] = *(const uint4*)(bp + 16);
      b1.q[1] = *(const uint4*)(bp + 24);

      // shared C init = -0.5*|b|^2 (same column per lane for both row tiles)
      const float ci = ldsBsq[nt * 16 + n16];
      v8f cin;
      #pragma unroll
      for (int r = 0; r < 8; ++r) cin[r] = ci;

      v8f c0 = __builtin_amdgcn_wmma_f32_16x16x32_bf16(false, a[0][0].v, false, b0.v,
                                                       (short)0, cin, false, false);
      v8f c1 = __builtin_amdgcn_wmma_f32_16x16x32_bf16(false, a[1][0].v, false, b0.v,
                                                       (short)0, cin, false, false);
      c0 = __builtin_amdgcn_wmma_f32_16x16x32_bf16(false, a[0][1].v, false, b1.v,
                                                   (short)0, c0, false, false);
      c1 = __builtin_amdgcn_wmma_f32_16x16x32_bf16(false, a[1][1].v, false, b1.v,
                                                   (short)0, c1, false, false);

      // C layout: lanes 0-15 vgpr r -> row r; lanes 16-31 vgpr r -> row 8+r.
      // argmin distance == argmax (a.b - 0.5|b|^2); strict > keeps low index.
      #pragma unroll
      for (int r = 0; r < 8; ++r) {
        if (c0[r] > maxv[r])     { maxv[r] = c0[r];     maxi[r] = col; }
        if (c1[r] > maxv[8 + r]) { maxv[8 + r] = c1[r]; maxi[8 + r] = col; }
      }
    }
    __syncthreads();
  }

  // ---- Cross-lane argmax within each 16-lane column group (tie -> low index)
  #pragma unroll
  for (int s = 0; s < 16; ++s) {
    float v = maxv[s]; int i = maxi[s];
    #pragma unroll
    for (int m = 1; m <= 8; m <<= 1) {
      float ov = __shfl_xor(v, m, 32);
      int   oi = __shfl_xor(i, m, 32);
      if (ov > v || (ov == v && oi < i)) { v = ov; i = oi; }
    }
    maxv[s] = v; maxi[s] = i;
  }

  // lanes 0 / 16 hold rows (tile*16 + half*8 + 0..7)
  if (n16 == 0) {
    #pragma unroll
    for (int s = 0; s < 16; ++s) {
      int rl  = (s >> 3) * 16 + half * 8 + (s & 7);
      int row = rowbase + rl;
      int ci  = maxi[s];
      out[(size_t)NELEMS + row] = (float)ci;     // indices output (as float)
      atomicAdd(&counts[ci], 1u);                // code-usage histogram
      widx[wid][rl] = ci;
    }
  }

  // ---- Gather fp32 codes, write straight-through output, accumulate MSE
  float sq = 0.f;
  for (int r = 0; r < 32; ++r) {
    int ci  = widx[wid][r];
    int row = rowbase + r;
    const float2 e = *(const float2*)(emb    + (size_t)ci  * DDIM + 2 * lane);
    const float2 x = *(const float2*)(inputs + (size_t)row * DDIM + 2 * lane);
    float dx = e.x - x.x, dy = e.y - x.y;
    float2 q; q.x = x.x + dx; q.y = x.y + dy;    // x + (quantized - x)
    *(float2*)(out + (size_t)row * DDIM + 2 * lane) = q;
    sq += dx * dx + dy * dy;
  }
  #pragma unroll
  for (int m = 1; m <= 16; m <<= 1) sq += __shfl_xor(sq, m, 32);
  if (lane == 0) partials[blockIdx.x * 8 + wid] = sq;
}

// ---------------------------------------------------------------------------
// Kernel 2: deterministic final reductions -> perplexity, vq_loss
// ---------------------------------------------------------------------------
__launch_bounds__(1024)
__global__ void vq_finalize(const uint32_t* __restrict__ counts,
                            const float* __restrict__ partials,
                            float* __restrict__ out) {
  __shared__ float red[1024];
  const int t = threadIdx.x;

  float s = 0.f;
  #pragma unroll
  for (int j = 0; j < 4; ++j) s += partials[t * 4 + j];
  red[t] = s;
  __syncthreads();
  for (int off = 512; off > 0; off >>= 1) {
    if (t < off) red[t] += red[t + off];
    __syncthreads();
  }
  float loss_sum = red[0];
  __syncthreads();

  float p = (float)counts[t] * (1.0f / (float)NROWS);
  red[t] = p * logf(p + EPSF);
  __syncthreads();
  for (int off = 512; off > 0; off >>= 1) {
    if (t < off) red[t] += red[t + off];
    __syncthreads();
  }

  if (t == 0) {
    out[(size_t)NELEMS + NROWS]     = expf(-red[0]);                    // perplexity
    out[(size_t)NELEMS + NROWS + 1] = 1.25f * loss_sum / (float)NELEMS; // vq_loss
  }
}

// ---------------------------------------------------------------------------
extern "C" void kernel_launch(void* const* d_in, const int* in_sizes, int n_in,
                              void* d_out, int out_size, void* d_ws, size_t ws_size,
                              hipStream_t stream) {
  (void)in_sizes; (void)n_in; (void)out_size; (void)ws_size;
  const float* inputs = (const float*)d_in[0];
  const float* emb    = (const float*)d_in[1];
  uint8_t* ws = (uint8_t*)d_ws;
  uint32_t* embbf  = (uint32_t*)(ws + WS_BF16);
  float*    bsqh   = (float*)   (ws + WS_BSQ);
  uint32_t* counts = (uint32_t*)(ws + WS_CNT);
  float*    parts  = (float*)   (ws + WS_PART);
  float*    out    = (float*)d_out;

  vq_prep    <<<KCODES / 256, 256, 0, stream>>>(emb, embbf, bsqh, counts);
  vq_main    <<<NROWS / 256, 256, 0, stream>>>(inputs, emb, embbf, bsqh,
                                               counts, parts, out);
  vq_finalize<<<1, 1024, 0, stream>>>(counts, parts, out);
}
